// PartQuery_80341658239242
// MI455X (gfx1250) — matrix operand order repaired and checked by reading, not compiled, not run on hardware
//
#include <hip/hip_runtime.h>
#include <hip/hip_bf16.h>
#include <math.h>

typedef __bf16 bf16_t;
typedef __attribute__((ext_vector_type(16))) __bf16 v16bf;
typedef __attribute__((ext_vector_type(8)))  float  v8f;
typedef __attribute__((ext_vector_type(4)))  unsigned int u32x4;
typedef __attribute__((ext_vector_type(8)))  int i32x8;
typedef __attribute__((ext_vector_type(4)))  int i32x4;

union ABFrag { v16bf f; uint4 q[2]; };

static __device__ __forceinline__ bf16_t f2bf(float x) {
  unsigned u = __float_as_uint(x);
  u += 0x7FFFu + ((u >> 16) & 1u);           // round-to-nearest-even
  unsigned short hs = (unsigned short)(u >> 16);
  return __builtin_bit_cast(bf16_t, hs);
}

static __device__ __forceinline__ v8f wmma_bf16(const ABFrag& a, const ABFrag& b, v8f c) {
  return __builtin_amdgcn_wmma_f32_16x16x32_bf16(false, a.f, false, b.f, (short)0, c, false, false);
}

// ---- TDM: load a 2D tile (128 rows x 32 bf16, row stride 768 elems) global->LDS ----
// LDS padding: 16 DWORDs (one 64B row) then +4 DWORDs -> effective row stride 80B (LDT=40).
static __device__ __forceinline__ void tdm_load_tile(unsigned lds_byte_off,
                                                     const bf16_t* gsrc) {
  unsigned long long ga = (unsigned long long)(size_t)gsrc;
  u32x4 g0;
  g0[0] = 1u;                                    // count=1, load, user mode
  g0[1] = lds_byte_off;                          // lds_addr
  g0[2] = (unsigned)ga;                          // global_addr[31:0]
  g0[3] = (unsigned)((ga >> 32) & 0x01FFFFFFull) // global_addr[56:32]
        | (2u << 30);                            // type = 2 ("image")
  i32x8 g1;
  g1[0] = (1 << 16)      // data_size = 1 (2 bytes)
        | (1 << 20)      // pad_enable
        | (3 << 22)      // pad_interval: 16 DWORDs (one 64B tile row)
        | (3 << 25);     // pad_amount: 4 DWORDs (16B)
  g1[1] = (int)(768u << 16);        // tensor_dim0[15:0] = 768 (bits 63:48)
  g1[2] = 0;                        // tensor_dim0[31:16]=0 | tensor_dim1[15:0]=0
  g1[3] = 16 | (32 << 16);          // tensor_dim1[31:16]=16 (dim1=1<<20) | tile_dim0=32
  g1[4] = 128;                      // tile_dim1=128, tile_dim2=0
  g1[5] = 768;                      // tensor_dim0_stride[31:0] = 768 elems
  g1[6] = 0;                        // stride0 hi | stride1 lo
  g1[7] = 0;
  i32x4 gz; gz[0] = 0; gz[1] = 0; gz[2] = 0; gz[3] = 0;  // dims 2..4 unused
  asm volatile("tensor_load_to_lds %0, %1, %2, %3"
               :: "s"(g0), "s"(g1), "s"(gz), "s"(gz)
               : "memory");
}

static __device__ __forceinline__ void wait_tensorcnt0() {
#if __has_builtin(__builtin_amdgcn_s_wait_tensorcnt)
  __builtin_amdgcn_s_wait_tensorcnt(0);
#else
  asm volatile("s_wait_tensorcnt 0x0" ::: "memory");
#endif
}

// ---------------- prep: x = bf16(origx + sinusoidal_pe) ----------------
__global__ __launch_bounds__(256) void pq_prep_x(const float* __restrict__ ox,
                                                 bf16_t* __restrict__ xbf, long total) {
  long i = (long)blockIdx.x * blockDim.x + threadIdx.x;
  const long stride = (long)gridDim.x * blockDim.x;
  const float kfreq = -9.210340372f / 768.0f;   // -ln(10000)/768
  for (; i < total; i += stride) {
    int  c = (int)(i % 768);
    long m = i / 768;
    int  p = (int)(m % 196);
    float ang = (float)p * __expf((float)(c & ~1) * kfreq);
    float pe  = (c & 1) ? __cosf(ang) : __sinf(ang);
    xbf[i] = f2bf(ox[i] + pe);
  }
}

// ---------------- prep: weight bf16 conversion ----------------
__global__ __launch_bounds__(256) void pq_prep_w(const float* __restrict__ wk,
                                                 const float* __restrict__ wv,
                                                 bf16_t* __restrict__ wkbf,
                                                 bf16_t* __restrict__ wvbf) {
  int i = blockIdx.x * 256 + threadIdx.x;
  if (i < 768 * 768) { wkbf[i] = f2bf(wk[i]); wvbf[i] = f2bf(wv[i]); }
}

// ---------------- prep: q = query @ W_q^T (32x768, tiny) ----------------
__global__ __launch_bounds__(256) void pq_prep_q(const float* __restrict__ query,
                                                 const float* __restrict__ wq,
                                                 bf16_t* __restrict__ qbf) {
  int g = blockIdx.x * 256 + threadIdx.x;
  if (g >= 32 * 768) return;
  int qq = g / 768, o = g % 768;
  const float* qr = query + (size_t)qq * 768;
  const float* wr = wq + (size_t)o * 768;
  float s = 0.f;
  for (int i = 0; i < 768; i += 4)
    s += qr[i]*wr[i] + qr[i+1]*wr[i+1] + qr[i+2]*wr[i+2] + qr[i+3]*wr[i+3];
  qbf[g] = f2bf(s);
}

// ---------------- k = x @ W_k^T : WMMA bf16 GEMM, TDM double-buffered ----------------
#define BM 128
#define BN 128
#define LDT 40   // padded LDS row stride (bf16 elems) == TDM pad layout

__global__ __launch_bounds__(256) void pq_gemm_k(const bf16_t* __restrict__ X,  // (50176,768)
                                                 const bf16_t* __restrict__ W,  // (768,768)
                                                 bf16_t* __restrict__ C) {      // (50176,768)
  __shared__ __attribute__((aligned(16))) bf16_t smem[4 * BM * LDT]; // [buf][A|B]
  const int m0 = blockIdx.x * BM;
  const int n0 = blockIdx.y * BN;
  const int t = threadIdx.x;
  const int wave = t >> 5, lane = t & 31;
  const int half = lane >> 4, lr = lane & 15;
  const int wm = (wave & 3) * 32;   // 2 m-subtiles per wave
  const int wn = (wave >> 2) * 64;  // 4 n-subtiles per wave
  const unsigned ldsBase = (unsigned)(size_t)(void*)&smem[0];
  const unsigned bufBytes = (unsigned)(2 * BM * LDT * sizeof(bf16_t)); // 20480
  const unsigned abBytes  = (unsigned)(BM * LDT * sizeof(bf16_t));     // 10240
  const bool issuer = (__builtin_amdgcn_readfirstlane(t) < 32);        // scalar branch: wave 0

  const v8f vzero = {};
  v8f acc[2][4];
  #pragma unroll
  for (int i = 0; i < 2; i++)
    #pragma unroll
    for (int j = 0; j < 4; j++) acc[i][j] = vzero;

  if (issuer) {                                   // prologue: tile 0 -> buffer 0
    tdm_load_tile(ldsBase, X + (size_t)m0 * 768);
    tdm_load_tile(ldsBase + abBytes, W + (size_t)n0 * 768);
  }

  for (int kt = 0; kt < 24; ++kt) {
    wait_tensorcnt0();                            // no-op for non-issuing waves
    __syncthreads();                              // tile kt visible to all
    if (kt + 1 < 24 && issuer) {                  // prefetch tile kt+1 into other buffer
      const unsigned bo = ldsBase + ((unsigned)((kt + 1) & 1)) * bufBytes;
      tdm_load_tile(bo, X + (size_t)m0 * 768 + (kt + 1) * 32);
      tdm_load_tile(bo + abBytes, W + (size_t)n0 * 768 + (kt + 1) * 32);
    }
    const bf16_t* Abuf = smem + (size_t)(kt & 1) * (2 * BM * LDT);
    const bf16_t* Bbuf = Abuf + BM * LDT;

    ABFrag a[2], b[4];
    #pragma unroll
    for (int i = 0; i < 2; i++) {    // A 16x32: halves hold K {0..7,16..23} / {8..15,24..31}
      const bf16_t* p = Abuf + (wm + i * 16 + lr) * LDT + half * 8;
      a[i].q[0] = *(const uint4*)p;
      a[i].q[1] = *(const uint4*)(p + 16);
    }
    #pragma unroll
    for (int j = 0; j < 4; j++) {    // B 32x16: lane=col, halves hold K {0..7,16..23} / {8..15,24..31}
      const bf16_t* p = Bbuf + (wn + j * 16 + lr) * LDT + half * 8;
      b[j].q[0] = *(const uint4*)p;
      b[j].q[1] = *(const uint4*)(p + 16);
    }
    #pragma unroll
    for (int i = 0; i < 2; i++)
      #pragma unroll
      for (int j = 0; j < 4; j++)
        acc[i][j] = wmma_bf16(a[i], b[j], acc[i][j]);
    __syncthreads();                 // all reads of this buffer done before re-fill
  }

  #pragma unroll
  for (int i = 0; i < 2; i++)
    #pragma unroll
    for (int j = 0; j < 4; j++) {
      const int n = n0 + wn + j * 16 + lr;
      #pragma unroll
      for (int v = 0; v < 8; v++) {
        const int m = m0 + wm + i * 16 + half * 8 + v;
        C[(size_t)m * 768 + n] = f2bf(acc[i][j][v]);
      }
    }
}

// ---------------- attention + pool + argmax + gathered value, per (b,h) ----------------
__global__ __launch_bounds__(256) void pq_attn(const bf16_t* __restrict__ qbf,   // (32,768)
                                               const bf16_t* __restrict__ kall,  // (50176,768)
                                               const bf16_t* __restrict__ xbf,   // (50176,768)
                                               const bf16_t* __restrict__ wvbf,  // (768,768)
                                               float* __restrict__ out) {        // (256,32,768)
  const int bh = blockIdx.x;
  const int b = bh >> 3;
  const int h = bh & 7;
  __shared__ __attribute__((aligned(16))) float attnS[32][208];
  __shared__ float redV[32][8];
  __shared__ int   redI[32][8];
  __shared__ float poolMax[32];
  __shared__ int   poolIdx[32];

  const int t = threadIdx.x;
  const int wave = t >> 5, lane = t & 31;
  const int half = lane >> 4, lr = lane & 15;
  const v8f vzero = {};

  // --- stage 1: attn[32,196] = q_h (32x96) x k_h^T, WMMA, frags straight from global ---
  for (int u = wave; u < 26; u += 8) {         // 2 m-tiles x 13 n-tiles
    const int mt = u & 1;
    const int nt = u >> 1;
    const int kpos = nt * 16 + lr;
    v8f acc = vzero;
    #pragma unroll
    for (int ks = 0; ks < 3; ks++) {           // hd=96 -> 3 K-steps of 32
      ABFrag a, bf;
      const bf16_t* ap = qbf + (size_t)(mt * 16 + lr) * 768 + h * 96 + ks * 32 + half * 8;
      a.q[0] = *(const uint4*)ap;
      a.q[1] = *(const uint4*)(ap + 16);
      if (kpos < 196) {
        const bf16_t* bp = kall + (size_t)(b * 196 + kpos) * 768 + h * 96 + ks * 32 + half * 8;
        bf.q[0] = *(const uint4*)bp;
        bf.q[1] = *(const uint4*)(bp + 16);
      } else {
        bf.q[0] = make_uint4(0, 0, 0, 0);
        bf.q[1] = make_uint4(0, 0, 0, 0);
      }
      acc = wmma_bf16(a, bf, acc);
    }
    #pragma unroll
    for (int v = 0; v < 8; v++)
      attnS[mt * 16 + half * 8 + v][nt * 16 + lr] = acc[v];
  }
  __syncthreads();

  // --- stage 2: 3x3 avg pool (zero pad) on 14x14 grid; max + argmax per q ---
  {
    const int q = t >> 3, s = t & 7;
    float best = -3.4e38f; int bidx = 0;
    for (int p = s; p < 196; p += 8) {
      const int r = p / 14, c0 = p % 14;
      float sum = 0.f;
      #pragma unroll
      for (int dr = -1; dr <= 1; dr++) {
        const int rr = r + dr;
        if (rr < 0 || rr > 13) continue;
        #pragma unroll
        for (int dc = -1; dc <= 1; dc++) {
          const int cc = c0 + dc;
          if (cc < 0 || cc > 13) continue;
          sum += attnS[q][rr * 14 + cc];
        }
      }
      const float pl = sum * (1.0f / 9.0f);
      if (pl > best || (pl == best && p < bidx)) { best = pl; bidx = p; }
    }
    redV[q][s] = best; redI[q][s] = bidx;
  }
  __syncthreads();
  if (t < 32) {
    float best = redV[t][0]; int bidx = redI[t][0];
    for (int s = 1; s < 8; s++) {
      const float v = redV[t][s]; const int ii = redI[t][s];
      if (v > best || (v == best && ii < bidx)) { best = v; bidx = ii; }
    }
    poolMax[t] = best; poolIdx[t] = bidx;
  }
  __syncthreads();

  // --- stage 3: value rows only at argmax: (32 gathered x rows) x W_v_h^T, WMMA ---
  for (int u = wave; u < 12; u += 8) {         // 2 m-tiles x 6 n-tiles (hd_v=96)
    const int mt = u & 1;
    const int nt = u >> 1;
    const int q = mt * 16 + lr;
    const int d = nt * 16 + lr;
    const size_t arow = (size_t)(b * 196 + poolIdx[q]) * 768;  // per-lane gathered row
    const size_t brow = (size_t)(h * 96 + d) * 768;
    v8f acc = vzero;
    for (int ks = 0; ks < 24; ks++) {          // 768 = 24 K-steps of 32
      ABFrag a, bf;
      const bf16_t* ap = xbf + arow + ks * 32 + half * 8;
      a.q[0] = *(const uint4*)ap;
      a.q[1] = *(const uint4*)(ap + 16);
      const bf16_t* bp = wvbf + brow + ks * 32 + half * 8;
      bf.q[0] = *(const uint4*)bp;
      bf.q[1] = *(const uint4*)(bp + 16);
      acc = wmma_bf16(a, bf, acc);
    }
    #pragma unroll
    for (int v = 0; v < 8; v++) {
      const int qq = mt * 16 + half * 8 + v;
      out[((size_t)b * 32 + qq) * 768 + h * 96 + nt * 16 + lr] = poolMax[qq] * acc[v];
    }
  }
}

extern "C" void kernel_launch(void* const* d_in, const int* in_sizes, int n_in,
                              void* d_out, int out_size, void* d_ws, size_t ws_size,
                              hipStream_t stream) {
  const float* origx = (const float*)d_in[0];
  const float* query = (const float*)d_in[1];
  const float* Wq    = (const float*)d_in[2];
  const float* Wk    = (const float*)d_in[3];
  const float* Wv    = (const float*)d_in[4];
  float* out = (float*)d_out;

  const size_t Mrows = (size_t)256 * 196;  // 50176
  char* ws = (char*)d_ws;
  size_t off = 0;
  bf16_t* xbf  = (bf16_t*)(ws + off); off += Mrows * 768 * 2;
  bf16_t* wkbf = (bf16_t*)(ws + off); off += (size_t)768 * 768 * 2;
  bf16_t* wvbf = (bf16_t*)(ws + off); off += (size_t)768 * 768 * 2;
  bf16_t* qbf  = (bf16_t*)(ws + off); off += (size_t)32 * 768 * 2;
  bf16_t* kall = (bf16_t*)(ws + off); off += Mrows * 768 * 2;
  if (off > ws_size) return;  // insufficient scratch

  const long totalX = (long)Mrows * 768;
  pq_prep_x<<<4096, 256, 0, stream>>>(origx, xbf, totalX);
  pq_prep_w<<<(768 * 768 + 255) / 256, 256, 0, stream>>>(Wk, Wv, wkbf, wvbf);
  pq_prep_q<<<96, 256, 0, stream>>>(query, Wq, qbf);
  dim3 gg(392, 6);
  pq_gemm_k<<<gg, 256, 0, stream>>>(xbf, wkbf, kall);
  pq_attn<<<2048, 256, 0, stream>>>(qbf, kall, xbf, wvbf, out);
}
